// FOLDSNet_40870908788753
// MI455X (gfx1250) — matrix operand order repaired and checked by reading, not compiled, not run on hardware
//
#include <hip/hip_runtime.h>
#include <hip/hip_bf16.h>

typedef __attribute__((ext_vector_type(16))) _Float16 v16h;
typedef __attribute__((ext_vector_type(8)))  float    v8f;

#define BATCH  256
#define NPIX   49152
#define NRET   3072
#define NV1    6144
#define NIT    3072
#define NCLS   1000
#define NCLSP  1024   // padded for 16-wide tiling

__device__ __forceinline__ float sigmoidf(float x) {
    return 1.0f / (1.0f + __expf(-x));
}

// ---- WMMA fragment loaders (wave32, 16-bit, K-major rows) -------------------
// A matrix 16x32 (ISA 7.12.2): lane m=L&15, hi=L>>4.
// VGPR j (0..3): K = 2j + 8*hi {pair}; VGPR 4+j: K = 16 + 2j + 8*hi {pair}.
// dword index into row = (j&3) + 4*hi + 8*(j>>2)
__device__ __forceinline__ v16h load_fragA(const _Float16* __restrict__ row, int hi) {
    union { v16h h; unsigned int u[8]; } f;
    const unsigned int* p = (const unsigned int*)row;
#pragma unroll
    for (int j = 0; j < 8; ++j)
        f.u[j] = p[(j & 3) + 4 * hi + 8 * (j >> 2)];
    return f.h;
}

// B matrix 32x16: lane holds column n=L&15; same K order as A:
// dword index = (j&3) + 4*hi + 8*(j>>2).
__device__ __forceinline__ v16h load_fragB(const _Float16* __restrict__ row, int hi) {
    union { v16h h; unsigned int u[8]; } f;
    const unsigned int* p = (const unsigned int*)row;
#pragma unroll
    for (int j = 0; j < 8; ++j)
        f.u[j] = p[(j & 3) + 4 * hi + 8 * (j >> 2)];
    return f.h;
}

// ---- Stage 1: retina + LGN fused, write f16 r2 ------------------------------
__global__ void retina_lgn_kernel(const float* __restrict__ x,
                                  const float* __restrict__ w_ret,
                                  const float* __restrict__ b_ret,
                                  const float* __restrict__ swlgn,
                                  const float* __restrict__ b_lgn,
                                  const int*   __restrict__ pixel_map,
                                  _Float16*    __restrict__ r2h) {
    int t = blockIdx.x * blockDim.x + threadIdx.x;
    if (t >= BATCH * NRET) return;
    int b = t / NRET, n = t % NRET;
    const float* xb = x + (size_t)b * NPIX;
    float dot = 0.f;
#pragma unroll
    for (int k = 0; k < 16; ++k)
        dot = fmaf(xb[pixel_map[n * 16 + k]], w_ret[n * 16 + k], dot);
    float r1 = sigmoidf(dot - b_ret[n]);
    float r2 = sigmoidf(r1 * swlgn[n] - b_lgn[n]);
    r2h[(size_t)b * NRET + n] = (_Float16)r2;
}

// ---- Row sums ---------------------------------------------------------------
__global__ void w_rowsum_kernel(const float* __restrict__ w, float* __restrict__ out,
                                int N, int K) {
    int t = blockIdx.x * blockDim.x + threadIdx.x;
    if (t >= N) return;
    float s = 0.f;
    for (int k = 0; k < K; ++k) s += w[(size_t)t * K + k];
    out[t] = s;
}

__global__ void mask_rowsum_kernel(const int* __restrict__ m, float* __restrict__ rs,
                                   int N, int K) {
    int lane = threadIdx.x & 31;
    int wave = threadIdx.x >> 5;
    int row  = blockIdx.x * (blockDim.x >> 5) + wave;
    if (row >= N) return;
    const int* mr = m + (size_t)row * K;
    int s = 0;
    for (int k = lane; k < K; k += 32) s += mr[k];
#pragma unroll
    for (int off = 16; off; off >>= 1) s += __shfl_xor(s, off, 32);
    if (lane == 0) rs[row] = (float)s;
}

// ---- int32 mask -> f16 ------------------------------------------------------
__global__ void mask_to_h_kernel(const int* __restrict__ m, _Float16* __restrict__ h,
                                 long count) {
    long t = (long)blockIdx.x * blockDim.x + threadIdx.x;
    if (t < count) h[t] = (_Float16)(float)m[t];
}

// ---- W_cls f32 -> f16, zero-padded 1000 -> 1024 rows ------------------------
__global__ void wcls_to_h_kernel(const float* __restrict__ w, _Float16* __restrict__ h) {
    int t = blockIdx.x * blockDim.x + threadIdx.x;
    if (t >= NCLSP * NRET) return;
    int n = t / NRET, k = t % NRET;
    h[t] = (_Float16)((n < NCLS) ? w[(size_t)n * NRET + k] : 0.f);
}

// ---- WMMA GEMM: C[M=256, N] = A[256,K] x Bm[N,K]^T, fused epilogue ----------
// EPI==0: out_h[m,n] = f16( sigmoid(acc/rs[n] * sw[n] - bias[n]) )
// EPI==1: out_f[m,n] = acc + bias[n]   (store only n < n_valid, stride out_stride)
// grid: (N/512, BATCH/32), block: 256 (8 waves); each wave: 32(M) x 64(N) tile
// -> 2 A frags + 4 B frags feed 8 WMMAs per K-step (6 dwords/WMMA).
template <int EPI>
__global__ void gemm_wmma_kernel(const _Float16* __restrict__ A,
                                 const _Float16* __restrict__ Bm,
                                 const float* __restrict__ rs,
                                 const float* __restrict__ sw,
                                 const float* __restrict__ bias,
                                 _Float16* __restrict__ outH,
                                 float* __restrict__ outF,
                                 int K, int N, int n_valid, int out_stride) {
    const int lane = threadIdx.x & 31;
    const int wave = threadIdx.x >> 5;
    const int hi   = lane >> 4;
    const int ml   = lane & 15;
    const int bm   = blockIdx.y * 32;
    const int bn   = (blockIdx.x * 8 + wave) * 64;

    const _Float16* ar0 = A + (size_t)(bm +  0 + ml) * K;
    const _Float16* ar1 = A + (size_t)(bm + 16 + ml) * K;
    const _Float16* br0 = Bm + (size_t)(bn +  0 + ml) * K;
    const _Float16* br1 = Bm + (size_t)(bn + 16 + ml) * K;
    const _Float16* br2 = Bm + (size_t)(bn + 32 + ml) * K;
    const _Float16* br3 = Bm + (size_t)(bn + 48 + ml) * K;

    v8f acc[2][4];
#pragma unroll
    for (int mt = 0; mt < 2; ++mt)
#pragma unroll
        for (int t = 0; t < 4; ++t) acc[mt][t] = (v8f){};

    for (int kb = 0; kb < K; kb += 32) {
        __builtin_prefetch(ar0 + kb + 128, 0, 1);      // global_prefetch_b8
        __builtin_prefetch(br0 + kb + 128, 0, 1);
        __builtin_prefetch(br2 + kb + 128, 0, 1);
        v16h a0 = load_fragA(ar0 + kb, hi);
        v16h a1 = load_fragA(ar1 + kb, hi);
        v16h b0 = load_fragB(br0 + kb, hi);
        v16h b1 = load_fragB(br1 + kb, hi);
        v16h b2 = load_fragB(br2 + kb, hi);
        v16h b3 = load_fragB(br3 + kb, hi);
        acc[0][0] = __builtin_amdgcn_wmma_f32_16x16x32_f16(false, a0, false, b0,
                                                           (short)0, acc[0][0], false, false);
        acc[0][1] = __builtin_amdgcn_wmma_f32_16x16x32_f16(false, a0, false, b1,
                                                           (short)0, acc[0][1], false, false);
        acc[0][2] = __builtin_amdgcn_wmma_f32_16x16x32_f16(false, a0, false, b2,
                                                           (short)0, acc[0][2], false, false);
        acc[0][3] = __builtin_amdgcn_wmma_f32_16x16x32_f16(false, a0, false, b3,
                                                           (short)0, acc[0][3], false, false);
        acc[1][0] = __builtin_amdgcn_wmma_f32_16x16x32_f16(false, a1, false, b0,
                                                           (short)0, acc[1][0], false, false);
        acc[1][1] = __builtin_amdgcn_wmma_f32_16x16x32_f16(false, a1, false, b1,
                                                           (short)0, acc[1][1], false, false);
        acc[1][2] = __builtin_amdgcn_wmma_f32_16x16x32_f16(false, a1, false, b2,
                                                           (short)0, acc[1][2], false, false);
        acc[1][3] = __builtin_amdgcn_wmma_f32_16x16x32_f16(false, a1, false, b3,
                                                           (short)0, acc[1][3], false, false);
    }

    // D layout: VGPR i -> M = i + 8*hi, N = lane&15 (ISA 7.12.2)
#pragma unroll
    for (int t = 0; t < 4; ++t) {
        int n = bn + t * 16 + ml;
        if (EPI == 0) {
            float inv = 1.0f / rs[n];
            float scl = sw[n];
            float bi  = bias[n];
#pragma unroll
            for (int mt = 0; mt < 2; ++mt) {
#pragma unroll
                for (int i = 0; i < 8; ++i) {
                    int m = bm + mt * 16 + hi * 8 + i;
                    float v = sigmoidf(acc[mt][t][i] * inv * scl - bi);
                    outH[(size_t)m * N + n] = (_Float16)v;
                }
            }
        } else {
            if (n < n_valid) {
                float bi = bias[n];
#pragma unroll
                for (int mt = 0; mt < 2; ++mt) {
#pragma unroll
                    for (int i = 0; i < 8; ++i) {
                        int m = bm + mt * 16 + hi * 8 + i;
                        outF[(size_t)m * out_stride + n] = acc[mt][t][i] + bi;
                    }
                }
            }
        }
    }
}

// ---- Host orchestration -----------------------------------------------------
extern "C" void kernel_launch(void* const* d_in, const int* in_sizes, int n_in,
                              void* d_out, int out_size, void* d_ws, size_t ws_size,
                              hipStream_t stream) {
    const float* x         = (const float*)d_in[0];
    const float* w_ret     = (const float*)d_in[1];
    const float* b_ret     = (const float*)d_in[2];
    const float* w_lgn     = (const float*)d_in[3];
    const float* b_lgn     = (const float*)d_in[4];
    const float* w_v1      = (const float*)d_in[5];
    const float* b_v1      = (const float*)d_in[6];
    const float* w_it      = (const float*)d_in[7];
    const float* b_it      = (const float*)d_in[8];
    const float* W_cls     = (const float*)d_in[9];
    const float* b_cls     = (const float*)d_in[10];
    const int*   pixel_map = (const int*)d_in[11];
    const int*   m1        = (const int*)d_in[12];
    const int*   m2        = (const int*)d_in[13];

    char* base = (char*)d_ws;
    size_t off = 0;
    auto alloc = [&](size_t bytes) {
        char* p = base + off;
        off += (bytes + 255) & ~(size_t)255;
        return p;
    };
    _Float16* r2h   = (_Float16*)alloc((size_t)BATCH * NRET * 2);
    _Float16* r3h   = (_Float16*)alloc((size_t)BATCH * NV1  * 2);
    _Float16* r4h   = (_Float16*)alloc((size_t)BATCH * NIT  * 2);
    _Float16* m1h   = (_Float16*)alloc((size_t)NV1  * NRET * 2);
    _Float16* m2h   = (_Float16*)alloc((size_t)NIT  * NV1  * 2);
    _Float16* wclsh = (_Float16*)alloc((size_t)NCLSP * NRET * 2);
    float* rs1   = (float*)alloc((size_t)NV1 * 4);
    float* rs2   = (float*)alloc((size_t)NIT * 4);
    float* swlgn = (float*)alloc((size_t)NRET * 4);
    float* swv1  = (float*)alloc((size_t)NV1 * 4);
    float* swit  = (float*)alloc((size_t)NIT * 4);

    // Row sums of weight fields
    w_rowsum_kernel<<<(NRET + 255) / 256, 256, 0, stream>>>(w_lgn, swlgn, NRET, 16);
    w_rowsum_kernel<<<(NV1 + 255) / 256, 256, 0, stream>>>(w_v1, swv1, NV1, 32);
    w_rowsum_kernel<<<(NIT + 255) / 256, 256, 0, stream>>>(w_it, swit, NIT, 32);

    // Mask row sums (one wave per row)
    mask_rowsum_kernel<<<NV1 / 8, 256, 0, stream>>>(m1, rs1, NV1, NRET);
    mask_rowsum_kernel<<<NIT / 8, 256, 0, stream>>>(m2, rs2, NIT, NV1);

    // f16 conversions
    long c1 = (long)NV1 * NRET, c2 = (long)NIT * NV1;
    mask_to_h_kernel<<<(unsigned)((c1 + 255) / 256), 256, 0, stream>>>(m1, m1h, c1);
    mask_to_h_kernel<<<(unsigned)((c2 + 255) / 256), 256, 0, stream>>>(m2, m2h, c2);
    wcls_to_h_kernel<<<(NCLSP * NRET + 255) / 256, 256, 0, stream>>>(W_cls, wclsh);

    // Retina + LGN -> r2 (f16)
    retina_lgn_kernel<<<(BATCH * NRET + 255) / 256, 256, 0, stream>>>(
        x, w_ret, b_ret, swlgn, b_lgn, pixel_map, r2h);

    // V1:  r3 = sigmoid((r2 @ m1^T)/rs1 * swv1 - b_v1)
    gemm_wmma_kernel<0><<<dim3(NV1 / 512, BATCH / 32), 256, 0, stream>>>(
        r2h, m1h, rs1, swv1, b_v1, r3h, nullptr, NRET, NV1, NV1, NV1);

    // IT:  r4 = sigmoid((r3 @ m2^T)/rs2 * swit - b_it)
    gemm_wmma_kernel<0><<<dim3(NIT / 512, BATCH / 32), 256, 0, stream>>>(
        r3h, m2h, rs2, swit, b_it, r4h, nullptr, NV1, NIT, NIT, NIT);

    // Classifier: out = r4 @ W_cls^T + b_cls  (f32, stride 1000, n<1000 guarded)
    gemm_wmma_kernel<1><<<dim3(NCLSP / 512, BATCH / 32), 256, 0, stream>>>(
        r4h, wclsh, nullptr, nullptr, b_cls, nullptr, (float*)d_out,
        NRET, NCLSP, NCLS, NCLS);
}
